// BasicMultiTokenPrediction_44186623542099
// MI455X (gfx1250) — compile-verified, matrix operand order checked by reading, not run on hardware
//
#include <hip/hip_runtime.h>
#include <hip/hip_bf16.h>

// ---------------------------------------------------------------------------
// Types
// ---------------------------------------------------------------------------
typedef __attribute__((ext_vector_type(8)))  unsigned short u16x8;
typedef __attribute__((ext_vector_type(16))) __bf16         bf16x16;
typedef __attribute__((ext_vector_type(8)))  float          v8f;

struct BfPair { u16x8 lo, hi; };   // 32 bytes == bf16x16

__device__ __forceinline__ unsigned short f2bf(float f) {
  unsigned u = __float_as_uint(f);
  u += 0x7FFFu + ((u >> 16) & 1u);           // round-to-nearest-even
  return (unsigned short)(u >> 16);
}

// Low 32 bits of a generic pointer to __shared__ = wave-relative LDS offset
// (ISA 10.2: LDS aperture addresses truncate to addr[31:0]).
__device__ __forceinline__ unsigned lds_off(const void* p) {
  return (unsigned)(unsigned long long)p;
}

// GLOBAL_LOAD_ASYNC_TO_LDS_B128 (VGLOBAL op 98, tracked by ASYNCcnt).
// dst = per-lane LDS byte address VGPR, src = per-lane 64-bit global address.
__device__ __forceinline__ void async_ld_b128(unsigned dst, unsigned long long src) {
  asm volatile("global_load_async_to_lds_b128 %0, %1, off"
               :: "v"(dst), "v"(src) : "memory");
}

__device__ __forceinline__ void wait_async0() {
  asm volatile("s_wait_asynccnt 0x0" ::: "memory");
}

// ---------------------------------------------------------------------------
// fp32 -> bf16 conversion (weights), grid-stride
// ---------------------------------------------------------------------------
__global__ __launch_bounds__(256)
void mtp_cvt_bf16(const float* __restrict__ src, unsigned short* __restrict__ dst,
                  size_t n) {
  size_t i = (size_t)blockIdx.x * blockDim.x + threadIdx.x;
  size_t stride = (size_t)gridDim.x * blockDim.x;
  for (; i < n; i += stride) dst[i] = f2bf(src[i]);
}

// ---------------------------------------------------------------------------
// merged = concat(rms(h), rms(e)) in bf16.   One block per (b,p) row.
// ---------------------------------------------------------------------------
__global__ __launch_bounds__(256)
void mtp_rms_merge(const float* __restrict__ hsrc, long long hbs,
                   const float* __restrict__ x, int koff,
                   const float* __restrict__ rms_w,
                   unsigned short* __restrict__ merged) {
  const int r = blockIdx.x;                // 0..505
  const int b = r / 253, p = r % 253;
  const float* hrow = hsrc + (size_t)b * hbs + (size_t)p * 1024;
  const float* erow = x + ((size_t)(b * 256 + p + koff)) * 1024;

  __shared__ float red[2][8];
  __shared__ float inv[2];
  float ssh = 0.f, sse = 0.f;
  for (int i = threadIdx.x; i < 1024; i += 256) {
    float a = hrow[i]; ssh += a * a;
    float e = erow[i]; sse += e * e;
  }
  for (int off = 16; off; off >>= 1) {
    ssh += __shfl_down(ssh, off, 32);
    sse += __shfl_down(sse, off, 32);
  }
  const int wave = threadIdx.x >> 5, lane = threadIdx.x & 31;
  if (lane == 0) { red[0][wave] = ssh; red[1][wave] = sse; }
  __syncthreads();
  if (threadIdx.x == 0) {
    float t0 = 0.f, t1 = 0.f;
    for (int i = 0; i < 8; ++i) { t0 += red[0][i]; t1 += red[1][i]; }
    inv[0] = rsqrtf(t0 * (1.0f / 1024.0f) + 1e-6f);
    inv[1] = rsqrtf(t1 * (1.0f / 1024.0f) + 1e-6f);
  }
  __syncthreads();
  unsigned short* mrow = merged + (size_t)r * 2048;
  for (int i = threadIdx.x; i < 1024; i += 256) {
    float w = rms_w[i];
    mrow[i]        = f2bf(hrow[i] * inv[0] * w);
    mrow[1024 + i] = f2bf(erow[i] * inv[1] * w);
  }
}

// ---------------------------------------------------------------------------
// out = layernorm(a + b2) * w + bias.  One block per row of 1024.
// ---------------------------------------------------------------------------
__global__ __launch_bounds__(256)
void mtp_add_ln(const float* __restrict__ a, const float* __restrict__ b2,
                const float* __restrict__ w, const float* __restrict__ bi,
                float* __restrict__ outf, unsigned short* __restrict__ outb) {
  const int r = blockIdx.x;
  const float* ar = a  + (size_t)r * 1024;
  const float* br = b2 + (size_t)r * 1024;

  __shared__ float red[2][8];
  __shared__ float stat[2];                 // mean, inv_std
  float s = 0.f, ss = 0.f;
  for (int i = threadIdx.x; i < 1024; i += 256) {
    float v = ar[i] + br[i];
    s += v; ss += v * v;
  }
  for (int off = 16; off; off >>= 1) {
    s  += __shfl_down(s,  off, 32);
    ss += __shfl_down(ss, off, 32);
  }
  const int wave = threadIdx.x >> 5, lane = threadIdx.x & 31;
  if (lane == 0) { red[0][wave] = s; red[1][wave] = ss; }
  __syncthreads();
  if (threadIdx.x == 0) {
    float t0 = 0.f, t1 = 0.f;
    for (int i = 0; i < 8; ++i) { t0 += red[0][i]; t1 += red[1][i]; }
    float mean = t0 * (1.0f / 1024.0f);
    float var  = t1 * (1.0f / 1024.0f) - mean * mean;
    stat[0] = mean;
    stat[1] = rsqrtf(var + 1e-5f);
  }
  __syncthreads();
  const float mean = stat[0], is = stat[1];
  for (int i = threadIdx.x; i < 1024; i += 256) {
    float v = (ar[i] + br[i] - mean) * is * w[i] + bi[i];
    if (outf) outf[(size_t)r * 1024 + i] = v;
    if (outb) outb[(size_t)r * 1024 + i] = f2bf(v);
  }
}

// ---------------------------------------------------------------------------
// Generic bf16 WMMA GEMM:  C[M][N] = sum_k A[m][k] * W[n][k] (+ bias) (relu?)
// A: M x Kd bf16 row-major (activations), W: N x Kd bf16 row-major (weights).
//
// Block tile 128x128, K-step 32, 8 waves; wave tile 32x64 -> 8 WMMAs/K-step.
// Double-buffered LDS staged with global_load_async_to_lds_b128 (ASYNCcnt),
// one barrier per K-step.  Edge tiles handled by clamping staged rows (no
// EXEC predication in the hot loop); stores are guarded in the epilogue.
// ---------------------------------------------------------------------------
#define BM 128
#define BN 128
#define BK 32
#define LDS_S 40   // padded row stride in bf16 elements (80 B, 16B-aligned)

__global__ __launch_bounds__(256)
void mtp_gemm_bf16_wmma(const unsigned short* __restrict__ A,
                        const unsigned short* __restrict__ W,
                        const float* __restrict__ bias,
                        float* __restrict__ Cf, long long ldc,
                        unsigned short* __restrict__ Cb,
                        int M, int N, int Kd, int relu) {
  __shared__ __align__(16) unsigned short As[2][BM * LDS_S];
  __shared__ __align__(16) unsigned short Bs[2][BN * LDS_S];

  const int tid  = threadIdx.x;
  const int wave = tid >> 5;
  const int lane = tid & 31;
  const int g    = lane >> 4;     // half-wave select (K split)
  const int l16  = lane & 15;     // row (A) / col (B,C) within a 16-group
  const int wm   = wave >> 1;     // 0..3 -> 32-row strip
  const int wn   = wave & 1;      // 0..1 -> 64-col strip
  const long long blockM = (long long)blockIdx.y * BM;
  const long long blockN = (long long)blockIdx.x * BN;

  // ---- per-thread staging chunks: tiles are 128 rows x 64 B = 512 x 16B ----
  const int c0  = tid,        c1  = tid + 256;
  const int r0  = c0 >> 2,    r1  = c1 >> 2;          // staged row 0..127
  const int kb0 = (c0 & 3) << 4, kb1 = (c1 & 3) << 4; // byte offset in row

  const long long aRow0 = (blockM + r0 < M) ? blockM + r0 : (long long)M - 1;
  const long long aRow1 = (blockM + r1 < M) ? blockM + r1 : (long long)M - 1;
  const long long bRow0 = (blockN + r0 < N) ? blockN + r0 : (long long)N - 1;
  const long long bRow1 = (blockN + r1 < N) ? blockN + r1 : (long long)N - 1;

  unsigned long long gA0 = (unsigned long long)A + ((unsigned long long)aRow0 * Kd) * 2 + kb0;
  unsigned long long gA1 = (unsigned long long)A + ((unsigned long long)aRow1 * Kd) * 2 + kb1;
  unsigned long long gB0 = (unsigned long long)W + ((unsigned long long)bRow0 * Kd) * 2 + kb0;
  unsigned long long gB1 = (unsigned long long)W + ((unsigned long long)bRow1 * Kd) * 2 + kb1;

  unsigned lA0[2], lA1[2], lB0[2], lB1[2];
#pragma unroll
  for (int s = 0; s < 2; ++s) {
    lA0[s] = lds_off(&As[s][0]) + r0 * (LDS_S * 2) + kb0;
    lA1[s] = lds_off(&As[s][0]) + r1 * (LDS_S * 2) + kb1;
    lB0[s] = lds_off(&Bs[s][0]) + r0 * (LDS_S * 2) + kb0;
    lB1[s] = lds_off(&Bs[s][0]) + r1 * (LDS_S * 2) + kb1;
  }

  v8f acc[2][4] = {};
  const int nk = Kd / BK;

  // prologue: stage K-tile 0 into buffer 0
  async_ld_b128(lA0[0], gA0);  async_ld_b128(lA1[0], gA1);
  async_ld_b128(lB0[0], gB0);  async_ld_b128(lB1[0], gB1);
  gA0 += 64; gA1 += 64; gB0 += 64; gB1 += 64;   // BK bf16 = 64 B

  for (int kt = 0; kt < nk; ++kt) {
    const int cur = kt & 1;
    wait_async0();        // own async writes to buf[cur] done
    __syncthreads();      // everyone's writes visible; buf[cur^1] free

    if (kt + 1 < nk) {    // prefetch next K-tile into the other buffer
      const int nxt = cur ^ 1;
      async_ld_b128(lA0[nxt], gA0);  async_ld_b128(lA1[nxt], gA1);
      async_ld_b128(lB0[nxt], gB0);  async_ld_b128(lB1[nxt], gB1);
      gA0 += 64; gA1 += 64; gB0 += 64; gB1 += 64;
    }

    // ---- fragments: lane holds K chunks {g*8..+7} and {16+g*8..+7} ----
    const unsigned short* aT = &As[cur][0];
    const unsigned short* bT = &Bs[cur][0];

    bf16x16 af[2];
#pragma unroll
    for (int s = 0; s < 2; ++s) {
      const unsigned short* arow = aT + (wm * 32 + s * 16 + l16) * LDS_S;
      BfPair ap;
      ap.lo = *(const u16x8*)(arow + g * 8);
      ap.hi = *(const u16x8*)(arow + 16 + g * 8);
      af[s] = __builtin_bit_cast(bf16x16, ap);
    }
#pragma unroll
    for (int j = 0; j < 4; ++j) {
      const unsigned short* brow = bT + (wn * 64 + j * 16 + l16) * LDS_S;
      BfPair bp;
      bp.lo = *(const u16x8*)(brow + g * 8);
      bp.hi = *(const u16x8*)(brow + 16 + g * 8);
      bf16x16 bf = __builtin_bit_cast(bf16x16, bp);
#pragma unroll
      for (int s = 0; s < 2; ++s)
        acc[s][j] = __builtin_amdgcn_wmma_f32_16x16x32_bf16(
            false, af[s], false, bf, (short)0, acc[s][j], false, false);
    }
    // next iteration's barrier also protects buf[cur] from being restaged
  }

  // ---- epilogue: C layout n = lane%16, m = vgpr + 8*(lane/16) ----
#pragma unroll
  for (int j = 0; j < 4; ++j) {
    long long col = blockN + wn * 64 + j * 16 + l16;
    if (col >= N) continue;
    float bv = bias ? bias[col] : 0.0f;
#pragma unroll
    for (int s = 0; s < 2; ++s) {
#pragma unroll
      for (int r = 0; r < 8; ++r) {
        long long row = blockM + wm * 32 + s * 16 + r + 8 * g;
        if (row >= M) continue;
        float v = acc[s][j][r] + bv;
        if (relu) v = fmaxf(v, 0.0f);
        if (Cf) Cf[row * ldc + col] = v;
        if (Cb) Cb[row * (long long)N + col] = f2bf(v);
      }
    }
  }
}

// ---------------------------------------------------------------------------
// Host side
// ---------------------------------------------------------------------------
static inline unsigned ceil_div_u(long long a, long long b) {
  return (unsigned)((a + b - 1) / b);
}

extern "C" void kernel_launch(void* const* d_in, const int* in_sizes, int n_in,
                              void* d_out, int out_size, void* d_ws, size_t ws_size,
                              hipStream_t stream) {
  (void)in_sizes; (void)n_in; (void)out_size; (void)ws_size;

  constexpr int Bb = 2, T = 256, D = 1024, KDEP = 3, V = 50257, DFF = 2048;
  constexpr int P = T - KDEP;            // 253
  constexpr int M = Bb * P;              // 506

  const float* x          = (const float*)d_in[0];
  const float* embed_w    = (const float*)d_in[1];
  const float* rms_w      = (const float*)d_in[2];
  const float* proj_w     = (const float*)d_in[3];
  const float* proj_b     = (const float*)d_in[4];
  const float* attn_in_w  = (const float*)d_in[5];
  const float* attn_in_b  = (const float*)d_in[6];
  const float* attn_out_w = (const float*)d_in[7];
  const float* attn_out_b = (const float*)d_in[8];
  const float* ff1_w      = (const float*)d_in[9];
  const float* ff1_b      = (const float*)d_in[10];
  const float* ff2_w      = (const float*)d_in[11];
  const float* ff2_b      = (const float*)d_in[12];
  const float* ln1_w      = (const float*)d_in[13];
  const float* ln1_b      = (const float*)d_in[14];
  const float* ln2_w      = (const float*)d_in[15];
  const float* ln2_b      = (const float*)d_in[16];
  float* out = (float*)d_out;

  // ---- workspace layout ----
  char* ws = (char*)d_ws;
  size_t off = 0;
  auto alloc = [&](size_t bytes) -> char* {
    char* p = ws + off;
    off += (bytes + 255) & ~(size_t)255;
    return p;
  };
  unsigned short* emb_bf  = (unsigned short*)alloc((size_t)V * D * 2);
  unsigned short* proj_bf = (unsigned short*)alloc((size_t)KDEP * D * 2 * D * 2);
  unsigned short* wv_bf   = (unsigned short*)alloc((size_t)KDEP * D * D * 2);
  unsigned short* ao_bf   = (unsigned short*)alloc((size_t)KDEP * D * D * 2);
  unsigned short* ff1_bf  = (unsigned short*)alloc((size_t)KDEP * DFF * D * 2);
  unsigned short* ff2_bf  = (unsigned short*)alloc((size_t)KDEP * D * DFF * 2);
  unsigned short* merged  = (unsigned short*)alloc((size_t)M * 2 * D * 2);
  float*          p_f     = (float*)         alloc((size_t)M * D * 4);
  unsigned short* p_b     = (unsigned short*)alloc((size_t)M * D * 2);
  unsigned short* v_b     = (unsigned short*)alloc((size_t)M * D * 2);
  float*          sa_f    = (float*)         alloc((size_t)M * D * 4);
  float*          s1_f    = (float*)         alloc((size_t)M * D * 4);
  unsigned short* s1_b    = (unsigned short*)alloc((size_t)M * D * 2);
  unsigned short* ff_b    = (unsigned short*)alloc((size_t)M * DFF * 2);
  float*          ff2_f   = (float*)         alloc((size_t)M * D * 4);
  float*          h_f     = (float*)         alloc((size_t)M * D * 4);
  unsigned short* h_b     = (unsigned short*)alloc((size_t)M * D * 2);

  auto cvt = [&](const float* s, unsigned short* d, size_t n) {
    unsigned blocks = ceil_div_u((long long)n, 256);
    if (blocks > 8192u) blocks = 8192u;
    mtp_cvt_bf16<<<blocks, 256, 0, stream>>>(s, d, n);
  };

  // ---- weight conversions (idempotent, deterministic) ----
  cvt(embed_w, emb_bf, (size_t)V * D);
  cvt(proj_w,  proj_bf, (size_t)KDEP * D * 2 * D);
  for (int k = 0; k < KDEP; ++k)   // wv = attn_in_w[k][2D:]  (D x D slice)
    cvt(attn_in_w + (size_t)k * 3 * D * D + (size_t)2 * D * D,
        wv_bf + (size_t)k * D * D, (size_t)D * D);
  cvt(attn_out_w, ao_bf,  (size_t)KDEP * D * D);
  cvt(ff1_w,      ff1_bf, (size_t)KDEP * DFF * D);
  cvt(ff2_w,      ff2_bf, (size_t)KDEP * D * DFF);

  auto gemm = [&](const unsigned short* A, const unsigned short* W,
                  const float* bias, float* Cf, long long ldc,
                  unsigned short* Cb, int m, int n, int kd, int relu) {
    dim3 grid(ceil_div_u(n, BN), ceil_div_u(m, BM));
    mtp_gemm_bf16_wmma<<<grid, 256, 0, stream>>>(A, W, bias, Cf, ldc, Cb,
                                                 m, n, kd, relu);
  };

  for (int k = 0; k < KDEP; ++k) {
    // merged = [rms(h), rms(e)] in bf16
    const float* hsrc = (k == 0) ? x : h_f;
    long long hbs = (k == 0) ? (long long)T * D : (long long)P * D;
    mtp_rms_merge<<<M, 256, 0, stream>>>(hsrc, hbs, x, k + 1, rms_w, merged);

    // p = merged @ proj_w[k]^T + proj_b[k]
    gemm(merged, proj_bf + (size_t)k * D * 2 * D, proj_b + (size_t)k * D,
         p_f, D, p_b, M, D, 2 * D, 0);
    // v = p @ wv^T + bv
    gemm(p_b, wv_bf + (size_t)k * D * D,
         attn_in_b + (size_t)k * 3 * D + 2 * D,
         nullptr, 0, v_b, M, D, D, 0);
    // sa = v @ attn_out_w[k]^T + attn_out_b[k]
    gemm(v_b, ao_bf + (size_t)k * D * D, attn_out_b + (size_t)k * D,
         sa_f, D, nullptr, M, D, D, 0);
    // s1 = LN(p + sa)
    mtp_add_ln<<<M, 256, 0, stream>>>(p_f, sa_f, ln1_w + (size_t)k * D,
                                      ln1_b + (size_t)k * D, s1_f, s1_b);
    // ff = relu(s1 @ ff1^T + b1)
    gemm(s1_b, ff1_bf + (size_t)k * DFF * D, ff1_b + (size_t)k * DFF,
         nullptr, 0, ff_b, M, DFF, D, 1);
    // ff2 = ff @ ff2^T + b2
    gemm(ff_b, ff2_bf + (size_t)k * D * DFF, ff2_b + (size_t)k * D,
         ff2_f, D, nullptr, M, D, DFF, 0);
    // h = LN(s1 + ff2)
    mtp_add_ln<<<M, 256, 0, stream>>>(s1_f, ff2_f, ln2_w + (size_t)k * D,
                                      ln2_b + (size_t)k * D, h_f, h_b);
    // logits_k = h @ embed^T  -> out[b][p][k][:] (row stride KDEP*V)
    gemm(h_b, emb_bf, nullptr, out + (size_t)k * V, (long long)KDEP * V,
         nullptr, M, V, D, 0);
  }
}
